// MDTA_41394894799835
// MI455X (gfx1250) — matrix operand order repaired and checked by reading, not compiled, not run on hardware
//
#include <hip/hip_runtime.h>
#include <hip/hip_bf16.h>

typedef __attribute__((ext_vector_type(16))) __bf16 v16bf;
typedef __attribute__((ext_vector_type(8)))  float  v8f;

#define BATCH   8
#define CDIM    192
#define QKVC    576
#define HEADS   4
#define CH      48
#define HW      16384     // 128*128
#define NROWS   (BATCH*HEADS*CH)

union BF16Frag { v16bf v; unsigned int u[8]; };

// A-matrix fragment (16x32 bf16), row-major source, stride in elements.
// ISA 7.12.2: lanes 0-15 -> M=lane, K pairs {0..7}∪{16..23}; lanes 16-31 -> K {8..15}∪{24..31}
__device__ __forceinline__ v16bf frag_load_A(const __bf16* __restrict__ base, int stride) {
    const int lane = threadIdx.x & 31;
    const int m = lane & 15, hi = lane >> 4;
    const __bf16* row = base + (long)m * stride;
    BF16Frag f;
#pragma unroll
    for (int j = 0; j < 8; ++j) {
        int k = (j < 4) ? (8 * hi + 2 * j) : (16 + 8 * hi + 2 * (j - 4));
        f.u[j] = *reinterpret_cast<const unsigned int*>(row + k);
    }
    return f.v;
}

// B-matrix fragment (32x16 bf16) from K-contiguous (transposed) storage Bt[n][k].
// lanes 0-15: N=lane, K {0..7}∪{16..23}; lanes 16-31: K {8..15}∪{24..31}
__device__ __forceinline__ v16bf frag_load_Bt(const __bf16* __restrict__ base, int stride) {
    const int lane = threadIdx.x & 31;
    const int n = lane & 15, hi = lane >> 4;
    const __bf16* col = base + (long)n * stride;
    BF16Frag f;
#pragma unroll
    for (int j = 0; j < 8; ++j) {
        int k = (j < 4) ? (8 * hi + 2 * j) : (16 + 8 * hi + 2 * (j - 4));
        f.u[j] = *reinterpret_cast<const unsigned int*>(col + k);
    }
    return f.v;
}

__device__ __forceinline__ unsigned lds_off(const void* p) {
    // __shared__ object in generic AS: low 32 bits are the wave-relative LDS offset
    return (unsigned)(uintptr_t)p;
}

__device__ __forceinline__ void async_load_b128(unsigned loff, const void* gaddr) {
    // GLOBAL_LOAD_ASYNC_TO_LDS_B128: LDS[loff] <- MEM[gaddr], tracked by ASYNCcnt
    asm volatile("global_load_async_to_lds_b128 %0, %1, off"
                 :: "v"(loff), "v"(gaddr) : "memory");
}

__device__ __forceinline__ void wait_async0() {
    asm volatile("s_wait_asynccnt 0x0" ::: "memory");
}

__device__ __forceinline__ unsigned pack_bf16(float lo, float hi) {
    union { __bf16 h[2]; unsigned u; } p;
    p.h[0] = (__bf16)lo; p.h[1] = (__bf16)hi;
    return p.u;
}

__device__ __forceinline__ unsigned pack_bf16h(__bf16 lo, __bf16 hi) {
    union { __bf16 h[2]; unsigned u; } p;
    p.h[0] = lo; p.h[1] = hi;
    return p.u;
}

// ---------------- small setup kernels ----------------

// kv[b][o] = sum_f prior[b][f] * w_kernel[o][f]   (o in 0..383)
__global__ void k_kv(const float* __restrict__ prior, const float* __restrict__ w_kernel,
                     float* __restrict__ kv) {
    int b = blockIdx.x, o = threadIdx.x;
    const float* p = prior + b * CDIM;
    const float* w = w_kernel + o * CDIM;
    float a = 0.f;
    for (int f = 0; f < CDIM; ++f) a += p[f] * w[f];
    kv[b * 384 + o] = a;
}

// Wmod[b][o][c] = bf16(w_qkv[o][c]*kv1[b][c]); bias[b][o] = sum_c w_qkv[o][c]*kv2[b][c]
__global__ void k_wmod(const float* __restrict__ w_qkv, const float* __restrict__ kv,
                       __bf16* __restrict__ Wmod, float* __restrict__ bias) {
    int id = blockIdx.x * blockDim.x + threadIdx.x;     // 0..4607
    int b = id / QKVC, o = id % QKVC;
    const float* kv1 = kv + b * 384;
    const float* kv2 = kv1 + CDIM;
    const float* w = w_qkv + o * CDIM;
    __bf16* wm = Wmod + ((long)b * QKVC + o) * CDIM;
    float acc = 0.f;
    for (int c = 0; c < CDIM; ++c) {
        wm[c] = (__bf16)(w[c] * kv1[c]);
        acc += w[c] * kv2[c];
    }
    bias[b * QKVC + o] = acc;
}

// ---------------- tiled WMMA GEMM: C[b] = A[b](MxK) * B[b](KxN) (+bias) ----------------
// Double-buffered LDS; A tile via GLOBAL_LOAD_ASYNC_TO_LDS_B128 (ASYNCcnt),
// B tile converted/transposed through VGPRs with packed b32 LDS stores.
template <bool B_IS_F32, bool OUT_BF16, bool HAS_BIAS>
__global__ __launch_bounds__(256)
void k_gemm(const __bf16* __restrict__ A, const void* __restrict__ Bsrc,
            void* __restrict__ C, const float* __restrict__ bias,
            int M, int K, int N,
            long strideA, long strideB, long strideC, int strideBias) {
    __shared__ __bf16 sA[2][64][40];    // 80B row stride: 16B aligned for async B128 writes
    __shared__ __bf16 sBt[2][128][34];  // [n][k]; 17-dword row stride -> conflict-free reads

    const int bat = blockIdx.z;
    const int m0 = blockIdx.y * 64;
    const int n0 = blockIdx.x * 128;
    const __bf16* Ab = A + (long)bat * strideA;
    const float*  Bf = (const float*)Bsrc  + (long)bat * strideB;
    const __bf16* Bh = (const __bf16*)Bsrc + (long)bat * strideB;

    const int tid = threadIdx.x;
    const int wave = tid >> 5, lane = tid & 31;
    const int s = wave >> 1;      // M strip (0..3) -> rows m0+16*s
    const int nh = wave & 1;      // N half  -> cols n0+64*nh

    // per-thread staging roles
    const int arow = tid >> 2, acb = (tid & 3) * 8;        // A: 4 threads/row, 16B each
    const int bkp = tid >> 4, bnb = (tid & 15) * 8;        // B: k-pair 2*bkp, 8 cols

    v8f acc[4];
#pragma unroll
    for (int t = 0; t < 4; ++t)
#pragma unroll
        for (int e = 0; e < 8; ++e) acc[t][e] = 0.f;

    auto issue_tile = [&](int kc, int buf) {
        // A tile 64x32 bf16 -> LDS, async (one B128 per lane)
        async_load_b128(lds_off(&sA[buf][arow][acb]),
                        Ab + (long)(m0 + arow) * K + kc + acb);
        // B tile 32(k) x 128(n) -> sBt[n][k], two k-rows packed per b32 store
        if (B_IS_F32) {
            const float* s0 = Bf + (long)(kc + 2 * bkp) * N + n0 + bnb;
            const float* s1 = s0 + N;
            float4 a0 = *reinterpret_cast<const float4*>(s0);
            float4 a1 = *reinterpret_cast<const float4*>(s0 + 4);
            float4 b0 = *reinterpret_cast<const float4*>(s1);
            float4 b1 = *reinterpret_cast<const float4*>(s1 + 4);
            unsigned pk[8] = {
                pack_bf16(a0.x, b0.x), pack_bf16(a0.y, b0.y),
                pack_bf16(a0.z, b0.z), pack_bf16(a0.w, b0.w),
                pack_bf16(a1.x, b1.x), pack_bf16(a1.y, b1.y),
                pack_bf16(a1.z, b1.z), pack_bf16(a1.w, b1.w)};
#pragma unroll
            for (int i = 0; i < 8; ++i)
                *reinterpret_cast<unsigned*>(&sBt[buf][bnb + i][2 * bkp]) = pk[i];
        } else {
            const __bf16* s0 = Bh + (long)(kc + 2 * bkp) * N + n0 + bnb;
            const __bf16* s1 = s0 + N;
#pragma unroll
            for (int i = 0; i < 8; ++i)
                *reinterpret_cast<unsigned*>(&sBt[buf][bnb + i][2 * bkp]) =
                    pack_bf16h(s0[i], s1[i]);
        }
    };

    const int steps = K / 32;
    issue_tile(0, 0);
    for (int it = 0; it < steps; ++it) {
        wait_async0();          // own async A writes landed in LDS
        __syncthreads();        // everyone's tile (async + ds stores) visible
        if (it + 1 < steps) issue_tile((it + 1) * 32, (it + 1) & 1);  // overlap next tile
        const int buf = it & 1;
        v16bf af = frag_load_A(&sA[buf][16 * s][0], 40);
#pragma unroll
        for (int nt = 0; nt < 4; ++nt) {
            v16bf bfg = frag_load_Bt(&sBt[buf][64 * nh + 16 * nt][0], 34);
            acc[nt] = __builtin_amdgcn_wmma_f32_16x16x32_bf16(
                false, af, false, bfg, (short)0, acc[nt], false, false);
        }
        __syncthreads();        // reads of buf done before it is refilled next round
    }

    // C/D layout: VGPR r holds M=r+8*hi, N=lane&15
    const int nl = lane & 15, hi = lane >> 4;
#pragma unroll
    for (int nt = 0; nt < 4; ++nt)
#pragma unroll
        for (int r = 0; r < 8; ++r) {
            int row = m0 + 16 * s + r + 8 * hi;
            int col = n0 + 64 * nh + 16 * nt + nl;
            float v = acc[nt][r];
            if (HAS_BIAS) v += bias[bat * strideBias + row];
            long off = (long)bat * strideC + (long)row * N + col;
            if (OUT_BF16) ((__bf16*)C)[off] = (__bf16)v;
            else          ((float*)C)[off] = v;
        }
}

// ---------------- depthwise 3x3 + fused sum-of-squares (for q,k norms) ----------------
__global__ __launch_bounds__(256)
void k_dwconv(const __bf16* __restrict__ in, const float* __restrict__ wdw,
              __bf16* __restrict__ out, float* __restrict__ sumsq) {
    const int c = blockIdx.x, bat = blockIdx.y;
    const __bf16* src = in + ((long)bat * QKVC + c) * HW;
    __bf16* dst = out + ((long)bat * QKVC + c) * HW;
    float w[9];
#pragma unroll
    for (int i = 0; i < 9; ++i) w[i] = wdw[c * 9 + i];
    float local = 0.f;
    for (int p = threadIdx.x; p < HW; p += 256) {
        int y = p >> 7, x = p & 127;
        float a = 0.f;
#pragma unroll
        for (int ky = 0; ky < 3; ++ky) {
            int yy = y + ky - 1;
            if (yy < 0 || yy > 127) continue;
#pragma unroll
            for (int kx = 0; kx < 3; ++kx) {
                int xx = x + kx - 1;
                if (xx < 0 || xx > 127) continue;
                a += w[ky * 3 + kx] * (float)src[yy * 128 + xx];
            }
        }
        dst[p] = (__bf16)a;
        local += a * a;
    }
    if (c < 384) {   // q (0..191) and k (192..383) channels need norms
        __shared__ float red[256];
        red[threadIdx.x] = local;
        __syncthreads();
        for (int st = 128; st > 0; st >>= 1) {
            if (threadIdx.x < st) red[threadIdx.x] += red[threadIdx.x + st];
            __syncthreads();
        }
        if (threadIdx.x == 0) sumsq[bat * 384 + c] = red[0];
    }
}

// ---------------- attention scores: S[b,h] += scaled (q @ k^T) over a K chunk ----------------
__global__ __launch_bounds__(256)
void k_attn(const __bf16* __restrict__ qdw, const float* __restrict__ sumsq,
            const float* __restrict__ temperature, float* __restrict__ S) {
    const int kchunk = blockIdx.x, h = blockIdx.y, bat = blockIdx.z;
    const int wave = threadIdx.x >> 5, lane = threadIdx.x & 31;
    const __bf16* qb = qdw + ((long)(bat * QKVC + h * CH)) * HW;
    const __bf16* kb = qdw + ((long)(bat * QKVC + CDIM + h * CH)) * HW;

    v8f acc[3][3];
#pragma unroll
    for (int mi = 0; mi < 3; ++mi)
#pragma unroll
        for (int nj = 0; nj < 3; ++nj)
#pragma unroll
            for (int e = 0; e < 8; ++e) acc[mi][nj][e] = 0.f;

    const int kstart = kchunk * 2048 + wave * 256;
    for (int kk = 0; kk < 256; kk += 32) {
        const int kq = kstart + kk;
        v16bf af[3], bfg[3];
#pragma unroll
        for (int mi = 0; mi < 3; ++mi)
            af[mi] = frag_load_A(qb + (long)(16 * mi) * HW + kq, HW);   // q rows K-contiguous
#pragma unroll
        for (int nj = 0; nj < 3; ++nj)
            bfg[nj] = frag_load_Bt(kb + (long)(16 * nj) * HW + kq, HW); // k rows = B^T rows
#pragma unroll
        for (int mi = 0; mi < 3; ++mi)
#pragma unroll
            for (int nj = 0; nj < 3; ++nj)
                acc[mi][nj] = __builtin_amdgcn_wmma_f32_16x16x32_bf16(
                    false, af[mi], false, bfg[nj], (short)0, acc[mi][nj], false, false);
    }

    const float temp = temperature[h];
    const int nl = lane & 15, hi = lane >> 4;
#pragma unroll
    for (int nj = 0; nj < 3; ++nj) {
        int col = 16 * nj + nl;
        float sk = 1.0f / fmaxf(sqrtf(sumsq[bat * 384 + CDIM + h * CH + col]), 1e-12f);
#pragma unroll
        for (int mi = 0; mi < 3; ++mi)
#pragma unroll
            for (int r = 0; r < 8; ++r) {
                int row = 16 * mi + r + 8 * hi;
                float sq = 1.0f / fmaxf(sqrtf(sumsq[bat * 384 + h * CH + row]), 1e-12f);
                float v = acc[mi][nj][r] * sq * sk * temp;
                atomicAdd(&S[(((long)bat * HEADS + h) * CH + row) * CH + col], v);
            }
    }
}

// ---------------- softmax over rows of 48 (in place) ----------------
__global__ void k_softmax(float* __restrict__ S) {
    int r = blockIdx.x * blockDim.x + threadIdx.x;
    if (r >= NROWS) return;
    float* row = S + (long)r * CH;
    float m = -1e30f;
    for (int j = 0; j < CH; ++j) m = fmaxf(m, row[j]);
    float e[CH], sum = 0.f;
    for (int j = 0; j < CH; ++j) { e[j] = __expf(row[j] - m); sum += e[j]; }
    float inv = 1.0f / sum;
    for (int j = 0; j < CH; ++j) row[j] = e[j] * inv;
}

// ---------------- M2[b] = w_proj * blockdiag(attn_b)  (192x192 bf16) ----------------
__global__ void k_m2(const float* __restrict__ S, const float* __restrict__ wproj,
                     __bf16* __restrict__ M2) {
    int id = blockIdx.x * blockDim.x + threadIdx.x;
    if (id >= BATCH * CDIM * CDIM) return;
    int bat = id / (CDIM * CDIM);
    int rem = id % (CDIM * CDIM);
    int o = rem / CDIM, cp = rem % CDIM;
    int h = cp / CH, j = cp % CH;
    const float* attn = S + (((long)bat * HEADS + h) * CH) * CH;
    const float* wp = wproj + o * CDIM + h * CH;
    float a = 0.f;
    for (int i = 0; i < CH; ++i) a += wp[i] * attn[i * CH + j];
    M2[(long)bat * (CDIM * CDIM) + rem] = (__bf16)a;
}

extern "C" void kernel_launch(void* const* d_in, const int* in_sizes, int n_in,
                              void* d_out, int out_size, void* d_ws, size_t ws_size,
                              hipStream_t stream) {
    const float* x        = (const float*)d_in[0];
    const float* prior    = (const float*)d_in[1];
    const float* w_qkv    = (const float*)d_in[2];
    const float* w_dw     = (const float*)d_in[3];
    const float* w_proj   = (const float*)d_in[4];
    const float* w_kernel = (const float*)d_in[5];
    const float* temp     = (const float*)d_in[6];

    char* ws = (char*)d_ws;
    size_t off = 0;
    auto alloc = [&](size_t bytes) { char* p = ws + off; off += (bytes + 255) & ~size_t(255); return p; };

    float*  kv    = (float*)alloc(BATCH * 384 * 4);
    __bf16* Wmod  = (__bf16*)alloc((size_t)BATCH * QKVC * CDIM * 2);
    float*  bias  = (float*)alloc(BATCH * QKVC * 4);
    __bf16* qkv   = (__bf16*)alloc((size_t)BATCH * QKVC * HW * 2);
    __bf16* qdw   = (__bf16*)alloc((size_t)BATCH * QKVC * HW * 2);
    float*  sumsq = (float*)alloc(BATCH * 384 * 4);
    float*  S     = (float*)alloc((size_t)NROWS * CH * 4);
    __bf16* M2    = (__bf16*)alloc((size_t)BATCH * CDIM * CDIM * 2);

    // 1) kv = prior @ w_kernel^T
    k_kv<<<BATCH, 384, 0, stream>>>(prior, w_kernel, kv);
    // 2) fold modulation into weights + bias
    k_wmod<<<(BATCH * QKVC) / 256, 256, 0, stream>>>(w_qkv, kv, Wmod, bias);
    // 3) qkv = Wmod @ x + bias   (bf16 out)
    k_gemm<true, true, true><<<dim3(HW / 128, QKVC / 64, BATCH), 256, 0, stream>>>(
        Wmod, (const void*)x, (void*)qkv, bias, QKVC, CDIM, HW,
        (long)QKVC * CDIM, (long)CDIM * HW, (long)QKVC * HW, QKVC);
    // 4) depthwise 3x3 + q/k sum-of-squares
    k_dwconv<<<dim3(QKVC, BATCH), 256, 0, stream>>>(qkv, w_dw, qdw, sumsq);
    // 5) scores
    hipMemsetAsync(S, 0, (size_t)NROWS * CH * 4, stream);
    k_attn<<<dim3(8, HEADS, BATCH), 256, 0, stream>>>(qdw, sumsq, temp, S);
    // 6) softmax
    k_softmax<<<(NROWS + 255) / 256, 256, 0, stream>>>(S);
    // 7) fuse proj with attn: M2 = w_proj * blockdiag(attn)
    k_m2<<<(BATCH * CDIM * CDIM) / 256, 256, 0, stream>>>(S, w_proj, M2);
    // 8) final = M2 @ v   (fp32 out to d_out)
    k_gemm<false, false, false><<<dim3(HW / 128, CDIM / 64, BATCH), 256, 0, stream>>>(
        M2, (const void*)(qdw + (size_t)384 * HW), d_out, nullptr, CDIM, CDIM, HW,
        (long)CDIM * CDIM, (long)QKVC * HW, (long)CDIM * HW, 0);
}
